// GRUDecoder_31250182046581
// MI455X (gfx1250) — matrix operand order repaired and checked by reading, not compiled, not run on hardware
//
#include <hip/hip_runtime.h>
#include <stdint.h>

typedef __attribute__((ext_vector_type(16))) __bf16 v16bf;
typedef __attribute__((ext_vector_type(8)))  __bf16 v8bf;
typedef __attribute__((ext_vector_type(8)))  float  v8f;

#define B_  256
#define T_  128
#define H_  1024
#define E_  1024

// ---------------------------------------------------------------------------
// Prologue 1: one-hot -> token id, with SOS(0) shift.  tok[b*T + t] is the
// token FED at step t (t==0 -> SOS index 0, else argmax(enc[b][t-1])).
// ---------------------------------------------------------------------------
__global__ __launch_bounds__(128) void k_tok(const float* __restrict__ enc,
                                             int* __restrict__ tok)
{
    const int t = blockIdx.x, b = blockIdx.y;
    if (t == 0) { if (threadIdx.x == 0) tok[b * T_] = 0; return; }
    const float* row = enc + ((size_t)b * T_ + (t - 1)) * E_;
#pragma unroll
    for (int i = 0; i < 8; ++i) {
        const int e = threadIdx.x * 8 + i;
        if (row[e] > 0.5f) tok[b * T_ + t] = e;   // exactly one lane fires
    }
}

// ---------------------------------------------------------------------------
// Prologue 2: w_ih (3H x E) -> w_ih_T (E x 3H) f32 (contiguous per-token row).
// ---------------------------------------------------------------------------
__global__ __launch_bounds__(256) void k_transpose(const float* __restrict__ src,
                                                   float* __restrict__ dst)
{
    const size_t i = (size_t)blockIdx.x * 256 + threadIdx.x;  // over 3072*1024
    const int n = (int)(i >> 10);
    const int e = (int)(i & 1023);
    dst[(size_t)e * (3 * H_) + n] = src[i];
}

// ---------------------------------------------------------------------------
// Prologue 3: pack (N x K) f32 weight, used as B = W^T (K x N), into WMMA
// bf16 B-fragment order (16-bit B 32x16: lanes 0-15 K=0..7,16..23, lanes 16-31
// K=8..15,24..31, N = lane%16).  Fragment(n_tile,k_step) = 32 lanes x 16 bf16,
// contiguous -> one 32B load per lane in the GEMM loop.
// ---------------------------------------------------------------------------
__global__ __launch_bounds__(256) void k_pack(const float* __restrict__ src,
                                              __bf16* __restrict__ dst, int K)
{
    const size_t i  = (size_t)blockIdx.x * 256 + threadIdx.x;  // over N*K
    const int e     = (int)(i & 15);
    const int lane  = (int)((i >> 4) & 31);
    const size_t r  = i >> 9;
    const int Ks    = K >> 5;
    const int ks    = (int)(r % Ks);
    const int nt    = (int)(r / Ks);
    const int n     = nt * 16 + (lane & 15);
    const int k     = ks * 32 + ((e >> 3) * 16) + ((lane >> 4) * 8) + (e & 7);
    dst[i] = (__bf16)src[(size_t)n * K + k];
}

// Prologue 4: context (f32) -> bf16 h buffer 0
__global__ __launch_bounds__(256) void k_hinit(const float* __restrict__ src,
                                               __bf16* __restrict__ dst)
{
    const size_t i = (size_t)blockIdx.x * 256 + threadIdx.x;
    dst[i] = (__bf16)src[i];
}

// ---------------------------------------------------------------------------
// One GRU timestep.  Grid (64 j-blocks, 8 b-pairs), 1 wave each.
// Each wave: 2 M-tiles x {r,z,n} = 6 accumulators sharing 3 B fragments and
// 2 A fragments per k-step, software-pipelined (next k-step's fragments are
// loaded before the current WMMAs issue, so waits overlap compute).
// ---------------------------------------------------------------------------
__global__ __launch_bounds__(32) void k_gru_step(
    const __bf16* __restrict__ hbfIn, __bf16* __restrict__ hbfOut,
    const float* __restrict__ hprevF, long hprevStride,
    float* __restrict__ houtF, long houtStride,
    const float* __restrict__ wihT, const int* __restrict__ tok,
    const float* __restrict__ b_ih, const float* __restrict__ b_hh,
    const __bf16* __restrict__ whhP, int t)
{
    const int lane = threadIdx.x;
    const int half = lane >> 4, lr = lane & 15;
    const int jb = blockIdx.x;
    const int bb0 = blockIdx.y * 2;               // two b-blocks per wave

    v8f acc[2][3] = {};
    const __bf16* aRow0 = hbfIn + (size_t)(bb0 * 16 + lr) * H_;
    const __bf16* aRow1 = aRow0 + (size_t)16 * H_;
    const int nt[3] = { jb, jb + 64, jb + 128 };  // r / z / n tiles

    auto loadA = [&](const __bf16* row, int ks) -> v16bf {
        const int k0 = ks * 32 + 8 * half;
        v8bf a0 = *(const v8bf*)(row + k0);
        v8bf a1 = *(const v8bf*)(row + k0 + 16);
        v16bf A;
#pragma unroll
        for (int i = 0; i < 8; ++i) { A[i] = a0[i]; A[8 + i] = a1[i]; }
        return A;
    };
    auto loadB = [&](int g, int ks) -> v16bf {
        return *(const v16bf*)(whhP + ((size_t)(nt[g] * 32 + ks) * 32 + lane) * 16);
    };

    v16bf A0c = loadA(aRow0, 0), A1c = loadA(aRow1, 0);
    v16bf B0c = loadB(0, 0), B1c = loadB(1, 0), B2c = loadB(2, 0);

#pragma unroll 2
    for (int ks = 0; ks < 32; ++ks) {
        const int kn = (ks < 31) ? ks + 1 : 31;   // branchless prefetch index
        v16bf A0n = loadA(aRow0, kn), A1n = loadA(aRow1, kn);
        v16bf B0n = loadB(0, kn), B1n = loadB(1, kn), B2n = loadB(2, kn);

        acc[0][0] = __builtin_amdgcn_wmma_f32_16x16x32_bf16(false, A0c, false, B0c, (short)0, acc[0][0], false, false);
        acc[0][1] = __builtin_amdgcn_wmma_f32_16x16x32_bf16(false, A0c, false, B1c, (short)0, acc[0][1], false, false);
        acc[0][2] = __builtin_amdgcn_wmma_f32_16x16x32_bf16(false, A0c, false, B2c, (short)0, acc[0][2], false, false);
        acc[1][0] = __builtin_amdgcn_wmma_f32_16x16x32_bf16(false, A1c, false, B0c, (short)0, acc[1][0], false, false);
        acc[1][1] = __builtin_amdgcn_wmma_f32_16x16x32_bf16(false, A1c, false, B1c, (short)0, acc[1][1], false, false);
        acc[1][2] = __builtin_amdgcn_wmma_f32_16x16x32_bf16(false, A1c, false, B2c, (short)0, acc[1][2], false, false);

        A0c = A0n; A1c = A1n; B0c = B0n; B1c = B1n; B2c = B2n;
    }

    const int j = jb * 16 + lr;
    const float bihR = b_ih[j], bihZ = b_ih[j + H_], bihN = b_ih[j + 2 * H_];
    const float bhhR = b_hh[j], bhhZ = b_hh[j + H_], bhhN = b_hh[j + 2 * H_];
    // C layout: vgpr v, lanes 0-15 -> M=v ; lanes 16-31 -> M=v+8
#pragma unroll
    for (int mt = 0; mt < 2; ++mt) {
#pragma unroll
        for (int v = 0; v < 8; ++v) {
            const int b  = (bb0 + mt) * 16 + v + 8 * half;
            const int tk = tok[b * T_ + t];
            const float* g = wihT + (size_t)tk * (3 * H_) + j;   // gi gather
            const float ir = g[0]      + bihR;
            const float iz = g[H_]     + bihZ;
            const float in = g[2 * H_] + bihN;
            const float hr = acc[mt][0][v] + bhhR;
            const float hz = acc[mt][1][v] + bhhZ;
            const float hn = acc[mt][2][v] + bhhN;
            const float rg = 1.f / (1.f + __expf(-(ir + hr)));
            const float zg = 1.f / (1.f + __expf(-(iz + hz)));
            const float ng = tanhf(in + rg * hn);
            const float hp = hprevF[(size_t)b * hprevStride + j];
            const float h  = (1.f - zg) * ng + zg * hp;
            houtF[(size_t)b * houtStride + j] = h;      // f32 h_t
            hbfOut[(size_t)b * H_ + j] = (__bf16)h;     // bf16 A for next step
        }
    }
}

// ---------------------------------------------------------------------------
// Batched output projection + log_softmax, in place over d_out.
// Block = 8 waves; owns 16 (b,t)-rows x full E.  h tile staged to LDS (bf16);
// B fragments from global are double-buffered across the K loop.
// ---------------------------------------------------------------------------
__global__ __launch_bounds__(256) void k_out(float* __restrict__ out,
                                             const __bf16* __restrict__ woutP,
                                             const float* __restrict__ b_out)
{
    __shared__ __bf16 sH[16 * H_];      // 32 KB
    __shared__ float  sPart[16][8];
    __shared__ float  sMax[16];
    __shared__ float  sLog[16];

    const int tid = threadIdx.x;
    const size_t rowBase = (size_t)blockIdx.x * 16;

    for (int i = tid; i < 16 * H_; i += 256)
        sH[i] = (__bf16)out[rowBase * E_ + i];
    __syncthreads();

    const int wave = tid >> 5, lane = tid & 31;
    const int half = lane >> 4, lr = lane & 15;

    auto loadA = [&](int ks) -> v16bf {
        const int k0 = ks * 32 + 8 * half;
        v8bf a0 = *(const v8bf*)(sH + lr * H_ + k0);
        v8bf a1 = *(const v8bf*)(sH + lr * H_ + k0 + 16);
        v16bf A;
#pragma unroll
        for (int i = 0; i < 8; ++i) { A[i] = a0[i]; A[8 + i] = a1[i]; }
        return A;
    };
    auto loadB = [&](int t8, int ks) -> v16bf {
        const int ntl = wave * 8 + t8;
        return *(const v16bf*)(woutP + ((size_t)(ntl * 32 + ks) * 32 + lane) * 16);
    };

    v8f acc[8] = {};
    v16bf Bc[8];
#pragma unroll
    for (int t8 = 0; t8 < 8; ++t8) Bc[t8] = loadB(t8, 0);

#pragma unroll 2
    for (int ks = 0; ks < 32; ++ks) {
        const int kn = (ks < 31) ? ks + 1 : 31;
        v16bf A = loadA(ks);
        v16bf Bn[8];
#pragma unroll
        for (int t8 = 0; t8 < 8; ++t8) Bn[t8] = loadB(t8, kn);
#pragma unroll
        for (int t8 = 0; t8 < 8; ++t8)
            acc[t8] = __builtin_amdgcn_wmma_f32_16x16x32_bf16(false, A, false, Bc[t8], (short)0, acc[t8], false, false);
#pragma unroll
        for (int t8 = 0; t8 < 8; ++t8) Bc[t8] = Bn[t8];
    }

    float bias[8];
#pragma unroll
    for (int t8 = 0; t8 < 8; ++t8) bias[t8] = b_out[(wave * 8 + t8) * 16 + lr];

    // -------- pass 1: row max -------------------------------------------
    float lm[8];
#pragma unroll
    for (int v = 0; v < 8; ++v) {
        float m = -3.4e38f;
#pragma unroll
        for (int t8 = 0; t8 < 8; ++t8) m = fmaxf(m, acc[t8][v] + bias[t8]);
        m = fmaxf(m, __shfl_xor(m, 1));
        m = fmaxf(m, __shfl_xor(m, 2));
        m = fmaxf(m, __shfl_xor(m, 4));
        m = fmaxf(m, __shfl_xor(m, 8));
        lm[v] = m;
    }
    if (lr == 0) {
#pragma unroll
        for (int v = 0; v < 8; ++v) sPart[v + 8 * half][wave] = lm[v];
    }
    __syncthreads();
    if (tid < 16) {
        float m = sPart[tid][0];
        for (int w = 1; w < 8; ++w) m = fmaxf(m, sPart[tid][w]);
        sMax[tid] = m;
    }
    __syncthreads();

    // -------- pass 2: row sum(exp(x - max)) ------------------------------
#pragma unroll
    for (int v = 0; v < 8; ++v) {
        const float m = sMax[v + 8 * half];
        float s = 0.f;
#pragma unroll
        for (int t8 = 0; t8 < 8; ++t8) s += __expf(acc[t8][v] + bias[t8] - m);
        s += __shfl_xor(s, 1);
        s += __shfl_xor(s, 2);
        s += __shfl_xor(s, 4);
        s += __shfl_xor(s, 8);
        lm[v] = s;
    }
    __syncthreads();
    if (lr == 0) {
#pragma unroll
        for (int v = 0; v < 8; ++v) sPart[v + 8 * half][wave] = lm[v];
    }
    __syncthreads();
    if (tid < 16) {
        float s = 0.f;
        for (int w = 0; w < 8; ++w) s += sPart[tid][w];
        sLog[tid] = __logf(s);
    }
    __syncthreads();

    // -------- pass 3: logp = x - max - log(sum) --------------------------
#pragma unroll
    for (int v = 0; v < 8; ++v) {
        const int row = v + 8 * half;
        const float m = sMax[row], lg = sLog[row];
        const size_t g = (rowBase + row) * E_;
#pragma unroll
        for (int t8 = 0; t8 < 8; ++t8) {
            const int col = (wave * 8 + t8) * 16 + lr;
            out[g + col] = acc[t8][v] + bias[t8] - m - lg;
        }
    }
}

// ---------------------------------------------------------------------------
extern "C" void kernel_launch(void* const* d_in, const int* in_sizes, int n_in,
                              void* d_out, int out_size, void* d_ws, size_t ws_size,
                              hipStream_t stream)
{
    const float* ctx   = (const float*)d_in[0];   // (B,H)
    const float* enc   = (const float*)d_in[1];   // (B,T,E) one-hot
    const float* w_ih  = (const float*)d_in[2];   // (3H,E)
    const float* w_hh  = (const float*)d_in[3];   // (3H,H)
    const float* b_ih  = (const float*)d_in[4];   // (3H)
    const float* b_hh  = (const float*)d_in[5];   // (3H)
    const float* w_out = (const float*)d_in[6];   // (E,H)
    const float* b_out = (const float*)d_in[7];   // (E)
    float* out = (float*)d_out;                   // (B,T,E)

    uint8_t* w = (uint8_t*)d_ws;
    int*    tok   = (int*)w;     w += (size_t)B_ * T_ * 4;            // 128 KB
    float*  wihT  = (float*)w;   w += (size_t)E_ * 3 * H_ * 4;        // 12 MB
    __bf16* whhP  = (__bf16*)w;  w += (size_t)3 * H_ * H_ * 2;        // 6 MB
    __bf16* woutP = (__bf16*)w;  w += (size_t)E_ * H_ * 2;            // 2 MB
    __bf16* hbf0  = (__bf16*)w;  w += (size_t)B_ * H_ * 2;            // 0.5 MB
    __bf16* hbf1  = (__bf16*)w;                                       // 0.5 MB

    k_tok      <<<dim3(T_, B_), 128, 0, stream>>>(enc, tok);
    k_transpose<<<(3 * H_ * E_) / 256, 256, 0, stream>>>(w_ih, wihT);
    k_pack     <<<(3 * H_ * H_) / 256, 256, 0, stream>>>(w_hh, whhP, H_);
    k_pack     <<<(E_ * H_)     / 256, 256, 0, stream>>>(w_out, woutP, H_);
    k_hinit    <<<(B_ * H_)     / 256, 256, 0, stream>>>(ctx, hbf0);

    for (int t = 0; t < T_; ++t) {
        const __bf16* hin  = (t & 1) ? hbf1 : hbf0;
        __bf16*       hout = (t & 1) ? hbf0 : hbf1;
        const float*  hpf  = (t == 0) ? ctx : (out + (size_t)(t - 1) * E_);
        const long    hps  = (t == 0) ? (long)H_ : (long)T_ * E_;
        k_gru_step<<<dim3(64, 8), 32, 0, stream>>>(
            hin, hout, hpf, hps,
            out + (size_t)t * E_, (long)T_ * E_,
            wihT, tok, b_ih, b_hh, whhP, t);
    }

    k_out<<<(B_ * T_) / 16, 256, 0, stream>>>(out, woutP, b_out);
}
